// SemanticMemory_18640158065231
// MI455X (gfx1250) — compile-verified, matrix operand order checked by reading, not run on hardware
//
#include <hip/hip_runtime.h>
#include <hip/hip_bf16.h>
#include <math.h>

typedef __attribute__((ext_vector_type(16))) __bf16 v16bf;
typedef __attribute__((ext_vector_type(8)))  __bf16 v8bf;
typedef __attribute__((ext_vector_type(8)))  float   v8f;

#define B_ROWS 16384
#define DIM    2048
#define SP     64

__device__ __forceinline__ float gelu_exact(float v) {
    return 0.5f * v * (1.0f + erff(v * 0.70710678118654752f));
}

// ---------------- f32 -> bf16 convert with row repack ----------------
__global__ void cvt_kernel(const float* __restrict__ src, __bf16* __restrict__ dst,
                           int n, int srcCols, int dstStride) {
    int i = blockIdx.x * blockDim.x + threadIdx.x;
    if (i >= n) return;
    int r = i / srcCols, c = i - r * srcCols;
    dst[(size_t)r * dstStride + c] = (__bf16)src[i];
}

// ---------------- prototype l2norm (for sim) + raw transpose (for retrieve) ----------------
__global__ void proto_prep_kernel(const float* __restrict__ P, __bf16* __restrict__ PN,
                                  __bf16* __restrict__ PT) {
    __shared__ float red[256];
    int s = blockIdx.x;
    const float* p = P + (size_t)s * DIM;
    float acc = 0.f;
    for (int d = threadIdx.x; d < DIM; d += 256) { float v = p[d]; acc += v * v; }
    red[threadIdx.x] = acc; __syncthreads();
    for (int o = 128; o > 0; o >>= 1) {
        if (threadIdx.x < o) red[threadIdx.x] += red[threadIdx.x + o];
        __syncthreads();
    }
    float inv = 1.0f / fmaxf(sqrtf(red[0]), 1e-12f);
    for (int d = threadIdx.x; d < DIM; d += 256) {
        float v = p[d];
        PN[(size_t)s * DIM + d] = (__bf16)(v * inv);
        PT[(size_t)d * SP + s]  = (__bf16)v;
    }
}

// ---------------- per-prototype salience bias ----------------
__global__ void bias_kernel(const float* __restrict__ conf, const float* __restrict__ age,
                            const float* __restrict__ ev, float* __restrict__ biasv) {
    __shared__ float sm[SP];
    int s = threadIdx.x;
    float e = ev[s];
    sm[s] = e; __syncthreads();
    for (int o = 32; o > 0; o >>= 1) { if (s < o) sm[s] = fmaxf(sm[s], sm[s + o]); __syncthreads(); }
    float emax = sm[0];
    float freq = logf(e + 1.0f) / (logf(emax + 2.0f) + 1e-8f);
    biasv[s] = 0.2f * expf(-age[s] * (1.0f / 200.0f)) + 0.15f * freq + 0.1f * conf[s] + 0.09f;
}

// ---------------- fragment loaders ----------------
// 16 rows x 32 K of bf16; lane L: row = L&15, koff = (L>>4)*8
// elems 0..7 = K[koff..koff+7], elems 8..15 = K[16+koff..23+koff]
__device__ __forceinline__ v16bf load_frag(const __bf16* __restrict__ base, int ld,
                                           int row0, int k0, int lane) {
    int r  = lane & 15;
    int hi = (lane >> 4) << 3;
    const __bf16* p = base + (size_t)(row0 + r) * ld + (k0 + hi);
    union { v16bf v; v8bf h[2]; } u;
    u.h[0] = *(const v8bf*)(p);
    u.h[1] = *(const v8bf*)(p + 16);
    return u.v;
}

// LDS tile is row-major [rows][32]
__device__ __forceinline__ v16bf load_frag_lds(const __bf16* base, int row0, int lane) {
    int r  = lane & 15;
    int hi = (lane >> 4) << 3;
    const __bf16* p = base + (row0 + r) * 32 + hi;
    union { v16bf v; v8bf h[2]; } u;
    u.h[0] = *(const v8bf*)(p);
    u.h[1] = *(const v8bf*)(p + 16);
    return u.v;
}

// ---------------- epilogue (shared by both GEMM kernels) ----------------
__device__ __forceinline__ void gemm_epilogue(v8f acc[4][4], int mBase, int nBase, int lane,
                                              const float* __restrict__ bias,
                                              const float* __restrict__ xadd, int ldx,
                                              float* __restrict__ outF,
                                              __bf16* __restrict__ outH, int ldo,
                                              int do_gelu) {
    int nc = lane & 15;
    int mh = (lane >> 4) * 8;
#pragma unroll
    for (int i = 0; i < 4; ++i) {
#pragma unroll
        for (int j = 0; j < 4; ++j) {
            int col = nBase + j * 16 + nc;
            float bv = bias ? bias[col] : 0.0f;
#pragma unroll
            for (int v = 0; v < 8; ++v) {
                int row = mBase + i * 16 + mh + v;
                float val = acc[i][j][v] + bv;
                if (do_gelu) val = gelu_exact(val);
                if (outF) {
                    float xv = xadd ? xadd[(size_t)row * ldx + col] : 0.0f;
                    outF[(size_t)row * ldo + col] = val + xv;
                } else {
                    outH[(size_t)row * ldo + col] = (__bf16)val;
                }
            }
        }
    }
}

// ---------------- direct-register WMMA GEMM (small N or small K cases) ----------------
__global__ __launch_bounds__(256)
void gemm_bf16_kernel(const __bf16* __restrict__ A, int lda,
                      const __bf16* __restrict__ Bw, int ldb,
                      const float* __restrict__ bias,
                      const float* __restrict__ xadd, int ldx,
                      float* __restrict__ outF, __bf16* __restrict__ outH, int ldo,
                      int K, int wavesN, int do_gelu) {
    int lane = threadIdx.x & 31;
    int wave = threadIdx.x >> 5;
    int wN = wave % wavesN;
    int wM = wave / wavesN;
    int nBase = (blockIdx.x * wavesN + wN) * 64;
    int mBase = (blockIdx.y * (8 / wavesN) + wM) * 64;

    v8f acc[4][4] = {};
    for (int k = 0; k < K; k += 32) {
        v16bf a[4], b[4];
#pragma unroll
        for (int i = 0; i < 4; ++i) a[i] = load_frag(A, lda, mBase + 16 * i, k, lane);
#pragma unroll
        for (int j = 0; j < 4; ++j) b[j] = load_frag(Bw, ldb, nBase + 16 * j, k, lane);
#pragma unroll
        for (int i = 0; i < 4; ++i)
#pragma unroll
            for (int j = 0; j < 4; ++j)
                acc[i][j] = __builtin_amdgcn_wmma_f32_16x16x32_bf16(
                    false, a[i], false, b[j], (short)0, acc[i][j], false, false);
    }
    gemm_epilogue(acc, mBase, nBase, lane, bias, xadd, ldx, outF, outH, ldo, do_gelu);
}

// ---------------- LDS-staged, double-buffered async GEMM (big GEMMs) ----------------
// Block: 256 threads = 8 waves (2 M x 4 N), tile 128(M) x 256(N), K-step 32.
// A stage: 128x32 bf16 (8KB), B stage: 256x32 bf16 (16KB); double buffered = 48KB LDS.
// Each thread issues 2 (A) + 4 (B) GLOBAL_LOAD_ASYNC_TO_LDS_B128 per stage (6/wave).
__device__ __forceinline__ void issue_stage(const __bf16* __restrict__ A, int lda, int mBlock,
                                            const __bf16* __restrict__ Bw, int ldb, int nBlock,
                                            int k, __bf16* As, __bf16* Bs, int tid) {
#pragma unroll
    for (int c = 0; c < 2; ++c) {
        int cid = tid * 2 + c;                 // 512 chunks of 16B for A
        int row = cid >> 2, sub = (cid & 3) * 8;
        const __bf16* g = A + (size_t)(mBlock + row) * lda + k + sub;
        unsigned l = (unsigned)(size_t)(As + row * 32 + sub);
        asm volatile("global_load_async_to_lds_b128 %0, %1, off"
                     :: "v"(l), "v"(g) : "memory");
    }
#pragma unroll
    for (int c = 0; c < 4; ++c) {
        int cid = tid * 4 + c;                 // 1024 chunks of 16B for B
        int row = cid >> 2, sub = (cid & 3) * 8;
        const __bf16* g = Bw + (size_t)(nBlock + row) * ldb + k + sub;
        unsigned l = (unsigned)(size_t)(Bs + row * 32 + sub);
        asm volatile("global_load_async_to_lds_b128 %0, %1, off"
                     :: "v"(l), "v"(g) : "memory");
    }
}

__global__ __launch_bounds__(256)
void gemm_lds_kernel(const __bf16* __restrict__ A, int lda,
                     const __bf16* __restrict__ Bw, int ldb,
                     const float* __restrict__ bias,
                     const float* __restrict__ xadd, int ldx,
                     float* __restrict__ outF, __bf16* __restrict__ outH, int ldo,
                     int K, int do_gelu) {
    __shared__ __bf16 As[2][128 * 32];
    __shared__ __bf16 Bs[2][256 * 32];

    int tid  = threadIdx.x;
    int lane = tid & 31;
    int wave = tid >> 5;
    int wN = wave & 3;
    int wM = wave >> 2;
    int nBlock = blockIdx.x * 256;
    int mBlock = blockIdx.y * 128;
    int nBase = wN * 64;   // within block tile
    int mBase = wM * 64;

    int nK = K / 32;
    issue_stage(A, lda, mBlock, Bw, ldb, nBlock, 0, As[0], Bs[0], tid);

    v8f acc[4][4] = {};
    for (int ks = 0; ks < nK; ++ks) {
        int cur = ks & 1;
        if (ks + 1 < nK) {
            issue_stage(A, lda, mBlock, Bw, ldb, nBlock, (ks + 1) * 32,
                        As[1 - cur], Bs[1 - cur], tid);
            asm volatile("s_wait_asynccnt 0x6" ::: "memory");   // stage ks landed
        } else {
            asm volatile("s_wait_asynccnt 0x0" ::: "memory");
        }
        __syncthreads();   // all waves' async writes for stage ks visible

        v16bf a[4], b[4];
#pragma unroll
        for (int i = 0; i < 4; ++i) a[i] = load_frag_lds(As[cur], mBase + 16 * i, lane);
#pragma unroll
        for (int j = 0; j < 4; ++j) b[j] = load_frag_lds(Bs[cur], nBase + 16 * j, lane);
#pragma unroll
        for (int i = 0; i < 4; ++i)
#pragma unroll
            for (int j = 0; j < 4; ++j)
                acc[i][j] = __builtin_amdgcn_wmma_f32_16x16x32_bf16(
                    false, a[i], false, b[j], (short)0, acc[i][j], false, false);

        __syncthreads();   // done reading buf[cur] before it is refilled
    }
    gemm_epilogue(acc, mBlock + mBase, nBlock + nBase, lane,
                  bias, xadd, ldx, outF, outH, ldo, do_gelu);
}

// ---------------- per-row 1/||h|| ----------------
__global__ void hnorm_kernel(const __bf16* __restrict__ H, float* __restrict__ rnorm) {
    __shared__ float red[256];
    int row = blockIdx.x;
    const __bf16* h = H + (size_t)row * DIM;
    float acc = 0.f;
    for (int i = threadIdx.x; i < DIM; i += 256) { float v = (float)h[i]; acc += v * v; }
    red[threadIdx.x] = acc; __syncthreads();
    for (int o = 128; o > 0; o >>= 1) {
        if (threadIdx.x < o) red[threadIdx.x] += red[threadIdx.x + o];
        __syncthreads();
    }
    if (threadIdx.x == 0) rnorm[row] = 1.0f / fmaxf(sqrtf(red[0]), 1e-12f);
}

// ---------------- type weights: softmax(r @ Wr2^T + br2), wave per row ----------------
__global__ void tw_kernel(const __bf16* __restrict__ R, const float* __restrict__ Wr2,
                          const float* __restrict__ br2, float* __restrict__ TW) {
    int lane = threadIdx.x & 31, wave = threadIdx.x >> 5;
    int row = blockIdx.x * 8 + wave;
    const __bf16* r = R + (size_t)row * (DIM / 2);
    float a0 = 0, a1 = 0, a2 = 0, a3 = 0;
    for (int i = lane; i < DIM / 2; i += 32) {
        float rv = (float)r[i];
        a0 += rv * Wr2[i];
        a1 += rv * Wr2[(DIM / 2) + i];
        a2 += rv * Wr2[2 * (DIM / 2) + i];
        a3 += rv * Wr2[3 * (DIM / 2) + i];
    }
    for (int o = 16; o > 0; o >>= 1) {
        a0 += __shfl_xor(a0, o, 32); a1 += __shfl_xor(a1, o, 32);
        a2 += __shfl_xor(a2, o, 32); a3 += __shfl_xor(a3, o, 32);
    }
    if (lane == 0) {
        a0 += br2[0]; a1 += br2[1]; a2 += br2[2]; a3 += br2[3];
        float m = fmaxf(fmaxf(a0, a1), fmaxf(a2, a3));
        float e0 = expf(a0 - m), e1 = expf(a1 - m), e2 = expf(a2 - m), e3 = expf(a3 - m);
        float inv = 1.0f / (e0 + e1 + e2 + e3);
        float* t = TW + (size_t)row * 4;
        t[0] = e0 * inv; t[1] = e1 * inv; t[2] = e2 * inv; t[3] = e3 * inv;
    }
}

// ---------------- salience -> clip -> softmax/T over S=64, wave per row ----------------
__global__ void attn_kernel(const float* __restrict__ SIM, const float* __restrict__ rnorm,
                            const float* __restrict__ TW, const float* __restrict__ conf,
                            const float* __restrict__ biasv, __bf16* __restrict__ ATT) {
    int lane = threadIdx.x & 31, wave = threadIdx.x >> 5;
    int row = blockIdx.x * 8 + wave;
    const float* s = SIM + (size_t)row * SP;
    float rn = rnorm[row];
    int i0 = lane, i1 = lane + 32;
    float v0 = 0.45f * s[i0] * rn * conf[i0] * TW[row * 4 + (i0 >> 4)] + biasv[i0];
    float v1 = 0.45f * s[i1] * rn * conf[i1] * TW[row * 4 + (i1 >> 4)] + biasv[i1];
    float sal0 = fminf(fmaxf(v0, 0.f), 1.f) * (1.0f / 0.07f);
    float sal1 = fminf(fmaxf(v1, 0.f), 1.f) * (1.0f / 0.07f);
    float m = fmaxf(sal0, sal1);
    for (int o = 16; o > 0; o >>= 1) m = fmaxf(m, __shfl_xor(m, o, 32));
    float e0 = expf(sal0 - m), e1 = expf(sal1 - m);
    float sum = e0 + e1;
    for (int o = 16; o > 0; o >>= 1) sum += __shfl_xor(sum, o, 32);
    float inv = 1.0f / sum;
    ATT[(size_t)row * SP + lane]      = (__bf16)(e0 * inv);
    ATT[(size_t)row * SP + lane + 32] = (__bf16)(e1 * inv);
}

// ---------------- in-place layernorm on d_out ----------------
__global__ void ln_kernel(float* __restrict__ Y, const float* __restrict__ w,
                          const float* __restrict__ b) {
    __shared__ float rs[256], rs2[256];
    int row = blockIdx.x;
    float* y = Y + (size_t)row * DIM;
    float vals[DIM / 256];
    float s = 0, s2 = 0;
#pragma unroll
    for (int k = 0; k < DIM / 256; ++k) {
        float v = y[threadIdx.x + k * 256];
        vals[k] = v; s += v; s2 += v * v;
    }
    rs[threadIdx.x] = s; rs2[threadIdx.x] = s2; __syncthreads();
    for (int o = 128; o > 0; o >>= 1) {
        if (threadIdx.x < o) { rs[threadIdx.x] += rs[threadIdx.x + o]; rs2[threadIdx.x] += rs2[threadIdx.x + o]; }
        __syncthreads();
    }
    float mu  = rs[0] * (1.0f / DIM);
    float var = rs2[0] * (1.0f / DIM) - mu * mu;
    float inv = rsqrtf(var + 1e-5f);
#pragma unroll
    for (int k = 0; k < DIM / 256; ++k) {
        int c = threadIdx.x + k * 256;
        y[c] = w[c] * (vals[k] - mu) * inv + b[c];
    }
}

extern "C" void kernel_launch(void* const* d_in, const int* in_sizes, int n_in,
                              void* d_out, int out_size, void* d_ws, size_t ws_size,
                              hipStream_t stream) {
    (void)in_sizes; (void)n_in; (void)out_size; (void)ws_size;
    const float* x      = (const float*)d_in[0];
    const float* protos = (const float*)d_in[1];
    const float* conf   = (const float*)d_in[2];
    const float* age    = (const float*)d_in[3];
    const float* ev     = (const float*)d_in[4];
    const float* W_in   = (const float*)d_in[5];
    const float* b_in   = (const float*)d_in[6];
    const float* Wr1    = (const float*)d_in[7];
    const float* br1    = (const float*)d_in[8];
    const float* Wr2    = (const float*)d_in[9];
    const float* br2    = (const float*)d_in[10];
    const float* W_out  = (const float*)d_in[11];
    const float* b_out  = (const float*)d_in[12];
    const float* ln_w   = (const float*)d_in[13];
    const float* ln_b   = (const float*)d_in[14];
    float* out = (float*)d_out;

    char* w = (char*)d_ws;
    size_t off = 0;
    auto alloc = [&](size_t bytes) {
        char* p = w + off;
        off += bytes;
        off = (off + 255) & ~(size_t)255;
        return p;
    };
    __bf16* CB    = (__bf16*)alloc((size_t)B_ROWS * 2 * DIM * 2); // [x | retrieved] bf16, ld=4096
    __bf16* HB    = (__bf16*)alloc((size_t)B_ROWS * DIM * 2);
    __bf16* RB    = (__bf16*)alloc((size_t)B_ROWS * (DIM / 2) * 2);
    __bf16* WINB  = (__bf16*)alloc((size_t)DIM * DIM * 2);
    __bf16* WR1B  = (__bf16*)alloc((size_t)(DIM / 2) * DIM * 2);
    __bf16* WOUTB = (__bf16*)alloc((size_t)DIM * 2 * DIM * 2);
    __bf16* PNB   = (__bf16*)alloc((size_t)SP * DIM * 2);
    __bf16* PTB   = (__bf16*)alloc((size_t)DIM * SP * 2);
    float*  SIMB  = (float*)alloc((size_t)B_ROWS * SP * 4);
    float*  RN    = (float*)alloc((size_t)B_ROWS * 4);
    float*  TW    = (float*)alloc((size_t)B_ROWS * 4 * 4);
    __bf16* ATT   = (__bf16*)alloc((size_t)B_ROWS * SP * 2);
    float*  BV    = (float*)alloc(SP * 4);

    dim3 blk(256);
    // convert inputs/weights to bf16
    cvt_kernel<<<(B_ROWS * DIM + 255) / 256, blk, 0, stream>>>(x, CB, B_ROWS * DIM, DIM, 2 * DIM);
    cvt_kernel<<<(DIM * DIM + 255) / 256, blk, 0, stream>>>(W_in, WINB, DIM * DIM, DIM, DIM);
    cvt_kernel<<<((DIM / 2) * DIM + 255) / 256, blk, 0, stream>>>(Wr1, WR1B, (DIM / 2) * DIM, DIM, DIM);
    cvt_kernel<<<(DIM * 2 * DIM + 255) / 256, blk, 0, stream>>>(W_out, WOUTB, DIM * 2 * DIM, 2 * DIM, 2 * DIM);
    proto_prep_kernel<<<SP, blk, 0, stream>>>(protos, PNB, PTB);
    bias_kernel<<<1, SP, 0, stream>>>(conf, age, ev, BV);

    // h = x @ W_in^T   (LDS-staged async GEMM)
    gemm_lds_kernel<<<dim3(DIM / 256, B_ROWS / 128), blk, 0, stream>>>(
        CB, 2 * DIM, WINB, DIM, b_in, nullptr, 0, nullptr, HB, DIM, DIM, 0);
    // r = gelu(x @ Wr1^T)
    gemm_lds_kernel<<<dim3((DIM / 2) / 256, B_ROWS / 128), blk, 0, stream>>>(
        CB, 2 * DIM, WR1B, DIM, br1, nullptr, 0, nullptr, RB, DIM / 2, DIM, 1);
    hnorm_kernel<<<B_ROWS, blk, 0, stream>>>(HB, RN);
    tw_kernel<<<B_ROWS / 8, blk, 0, stream>>>(RB, Wr2, br2, TW);
    // raw_sim = h @ p_norm^T  (N=64 -> direct-register GEMM)
    gemm_bf16_kernel<<<dim3(1, B_ROWS / 512), blk, 0, stream>>>(
        HB, DIM, PNB, DIM, nullptr, nullptr, 0, SIMB, nullptr, SP, DIM, 1, 0);
    attn_kernel<<<B_ROWS / 8, blk, 0, stream>>>(SIMB, RN, TW, conf, BV, ATT);
    // retrieved = attn @ prototypes (K=64 -> direct-register GEMM) -> second half of combined
    gemm_bf16_kernel<<<dim3(DIM / 256, B_ROWS / 128), blk, 0, stream>>>(
        ATT, SP, PTB, SP, nullptr, nullptr, 0, nullptr, CB + DIM, 2 * DIM, SP, 4, 0);
    // y = gelu(combined @ W_out^T + b_out) + x -> d_out (f32)
    gemm_lds_kernel<<<dim3(DIM / 256, B_ROWS / 128), blk, 0, stream>>>(
        CB, 2 * DIM, WOUTB, 2 * DIM, b_out, x, DIM, out, nullptr, DIM, 2 * DIM, 1);
    // layernorm in place
    ln_kernel<<<B_ROWS, blk, 0, stream>>>(out, ln_w, ln_b);
}
